// Edge_attention_88098369176143
// MI455X (gfx1250) — matrix operand order repaired and checked by reading, not compiled, not run on hardware
//
#include <hip/hip_runtime.h>
#include <math.h>

// ---------------------------------------------------------------------------
// CDNA5 (gfx1250) edge-attention GNN. wave32. WMMA f16 (16x16x32, f32 acc).
// ---------------------------------------------------------------------------

typedef __attribute__((ext_vector_type(16))) _Float16 v16h;
typedef __attribute__((ext_vector_type(8)))  _Float16 v8h;
typedef __attribute__((ext_vector_type(8)))  float    v8f;

#define WMMA16(a, b, c) \
  __builtin_amdgcn_wmma_f32_16x16x32_f16(false, (a), false, (b), (short)0, (c), false, false)

// order-preserving float <-> uint mapping for atomic max on f32 (incl. negatives)
__device__ inline unsigned enc_ord(float f) {
  unsigned b = __float_as_uint(f);
  return (b & 0x80000000u) ? ~b : (b | 0x80000000u);
}
__device__ inline float dec_ord(unsigned u) {
  unsigned b = (u & 0x80000000u) ? (u ^ 0x80000000u) : ~u;
  return __uint_as_float(b);
}

__device__ inline void cvt8(const float* __restrict__ p, _Float16* o) {
  float4 a = *(const float4*)(p);
  float4 b = *(const float4*)(p + 4);
  o[0] = (_Float16)a.x; o[1] = (_Float16)a.y; o[2] = (_Float16)a.z; o[3] = (_Float16)a.w;
  o[4] = (_Float16)b.x; o[5] = (_Float16)b.y; o[6] = (_Float16)b.z; o[7] = (_Float16)b.w;
}

// ---------------------------------------------------------------------------
// Convert row-major f32 weights [Krows x Ncols] into f16 B-fragment layout.
// Fragment (c = K-chunk of 32, n = 16-col tile) stored as 32 lanes x 16 halfs:
//   out[((c*nTilesN + n)*32 + lane)*16 + j] = w[K][col]
//   col = 16n + (lane&15),  K = 32c + (lane>>4)*8 + (j&7) + (j>>3)*16     (cols >= Ncols -> 0)
// ---------------------------------------------------------------------------
__global__ void prep_wfrag_kernel(const float* __restrict__ w, int Krows, int Ncols,
                                  _Float16* __restrict__ out, int nTilesN) {
  int t = blockIdx.x * blockDim.x + threadIdx.x;
  int nFrags = (Krows >> 5) * nTilesN;
  if (t >= nFrags * 32) return;
  int lane = t & 31;
  int frag = t >> 5;
  int c = frag / nTilesN;
  int n = frag % nTilesN;
  int hi = lane >> 4;
  int col = (n << 4) + (lane & 15);
  _Float16* o = out + (size_t)(frag * 32 + lane) * 16;
  for (int j = 0; j < 16; ++j) {
    int K = (c << 5) + (hi << 3) + (j & 7) + ((j >> 3) << 4);
    float v = (col < Ncols) ? w[(size_t)K * Ncols + col] : 0.0f;
    o[j] = (_Float16)v;
  }
}

// ---------------------------------------------------------------------------
// Stage 1: per-edge MLP scores. One wave handles a 16-edge tile.
//   x = concat(feat[src], feat[dst]) : [16, 2F]
//   hmid = relu(x @ w1 + b1)         : [16, 128]   (8 col-tiles x KC1 wmma)
//   s    = leaky(hmid @ w2 + b2)     : [16, 4]     (4 wmma, cols 4..15 padded)
// A-frag (16x32 f16, ISA 7.12.2): lane l row M=l&15, halfs j:
//   j<8  -> K = 32c + (l>>4)*8 + j         (8-aligned run)
//   j>=8 -> K = 32c + 16 + (l>>4)*8 + j-8  (8-aligned run)
// ---------------------------------------------------------------------------
template <int KC1>
__global__ __launch_bounds__(256) void edge_mlp_kernel(
    const float* __restrict__ feat, int F,
    const _Float16* __restrict__ w1f, const float* __restrict__ b1,
    const _Float16* __restrict__ w2f, const float* __restrict__ b2,
    const int* __restrict__ src, const int* __restrict__ dst,
    float* __restrict__ scoreOut, unsigned* __restrict__ nmax, int E) {
  __shared__ __align__(32) _Float16 lds[8][16 * 128];  // per-wave 16x128 hidden tile

  int wib  = threadIdx.x >> 5;
  int lane = threadIdx.x & 31;
  int tile = blockIdx.x * (blockDim.x >> 5) + wib;
  int numTiles = (E + 15) >> 4;
  if (tile >= numTiles) return;  // uniform per wave -> EXEC stays all-ones

  int m  = lane & 15;
  int hi = lane >> 4;
  int e  = tile * 16 + m;
  if (e >= E) e = E - 1;  // clamp tail (results masked at store)
  int sn = src[e], dn = dst[e];

  // ---- gather A fragments (edge features, f32 -> f16) ----
  v16h a[KC1];
  for (int c = 0; c < KC1; ++c) {
    union { v16h v; _Float16 h[16]; } u;
    int K0 = 32 * c + hi * 8;
    int K1 = K0 + 16;
    const float* p0 = (K0 < F) ? (feat + (size_t)sn * F + K0)
                               : (feat + (size_t)dn * F + (K0 - F));
    const float* p1 = (K1 < F) ? (feat + (size_t)sn * F + K1)
                               : (feat + (size_t)dn * F + (K1 - F));
    cvt8(p0, u.h);
    cvt8(p1, u.h + 8);
    a[c] = u.v;
  }

  // ---- layer 1: 8 column tiles of 16, accumulate over KC1 chunks ----
  for (int n = 0; n < 8; ++n) {
    v8f acc = {};
    for (int c = 0; c < KC1; ++c) {
      v16h b = *(const v16h*)(w1f + (size_t)(((c << 3) + n) * 32 + lane) * 16);
      acc = WMMA16(a[c], b, acc);
    }
    // C layout: lane l holds N = l&15, rows M = (l>>4)*8 + r. bias + relu -> LDS f16
    union { v8f v; float f[8]; } ua;
    ua.v = acc;
    float bv = b1[(n << 4) + m];
    for (int r = 0; r < 8; ++r) {
      float vv = ua.f[r] + bv;
      vv = vv > 0.0f ? vv : 0.0f;
      lds[wib][(hi * 8 + r) * 128 + (n << 4) + m] = (_Float16)vv;
    }
  }

  // ---- layer 2: [16,128] @ [128,16-padded]; A-frags re-read from LDS ----
  // (wave-private LDS; DS ops are in-order within a wave, no barrier needed)
  v8f acc2 = {};
  for (int c2 = 0; c2 < 4; ++c2) {
    int K0 = 32 * c2 + hi * 8;
    union { v16h v; v8h h2[2]; } u2;
    u2.h2[0] = *(const v8h*)(&lds[wib][m * 128 + K0]);
    u2.h2[1] = *(const v8h*)(&lds[wib][m * 128 + K0 + 16]);
    v16h b = *(const v16h*)(w2f + (size_t)(c2 * 32 + lane) * 16);
    acc2 = WMMA16(u2.v, b, acc2);
  }

  // ---- epilogue: bias, leaky relu, store scores, atomic node-max ----
  if (m < 4) {  // only head columns 0..3 are real
    union { v8f v; float f[8]; } ua;
    ua.v = acc2;
    float bv = b2[m];
    for (int r = 0; r < 8; ++r) {
      int M  = hi * 8 + r;
      int ee = tile * 16 + M;
      if (ee < E) {
        float vv = ua.f[r] + bv;
        vv = vv > 0.0f ? vv : 0.01f * vv;  // LeakyReLU(0.01)
        scoreOut[(size_t)ee * 4 + m] = vv;
        int dnn = dst[ee];
        atomicMax(nmax + (size_t)dnn * 4 + m, enc_ord(vv));
      }
    }
  }
}

// ---------------------------------------------------------------------------
// Stage 2: e = exp(score - max[dst]) in place; atomic per-node sums.
// ---------------------------------------------------------------------------
__global__ void softmax_expsum_kernel(float* __restrict__ score,
                                      const int* __restrict__ dst,
                                      const unsigned* __restrict__ nmax,
                                      float* __restrict__ nsum, int E) {
  int t = blockIdx.x * blockDim.x + threadIdx.x;
  if (t >= E * 4) return;
  int e = t >> 2, hh = t & 3;
  int d = dst[e];
  float mx = dec_ord(nmax[(size_t)d * 4 + hh]);
  float ex = __expf(score[t] - mx);
  score[t] = ex;
  atomicAdd(nsum + (size_t)d * 4 + hh, ex);
}

// ---------------------------------------------------------------------------
// Stage 3: one wave per edge; fused both branches.
//   out[dst][h][0:64]   += h[src] * a_pca[h]
//   out[dst][h][64:128] += h[src] * a_pi[h]
// ---------------------------------------------------------------------------
__global__ __launch_bounds__(256) void aggregate_kernel(
    const float* __restrict__ hfeat,
    const float* __restrict__ exp_pca, const float* __restrict__ sum_pca,
    const float* __restrict__ exp_pi,  const float* __restrict__ sum_pi,
    const int* __restrict__ src, const int* __restrict__ dst,
    float* __restrict__ out, int E) {
  int wave = (blockIdx.x * blockDim.x + threadIdx.x) >> 5;
  int lane = threadIdx.x & 31;
  if (wave >= E) return;
  int e = __builtin_amdgcn_readfirstlane(wave);  // force scalar edge index
  int s = src[e], d = dst[e];
  float apca[4], api[4];
  for (int hh = 0; hh < 4; ++hh) {
    apca[hh] = exp_pca[(size_t)e * 4 + hh] / sum_pca[(size_t)d * 4 + hh];
    api[hh]  = exp_pi [(size_t)e * 4 + hh] / sum_pi [(size_t)d * 4 + hh];
  }
  float hv0 = hfeat[(size_t)s * 64 + lane];
  float hv1 = hfeat[(size_t)s * 64 + lane + 32];
  float* ob = out + (size_t)d * 512;  // [4 heads][128]
  for (int hh = 0; hh < 4; ++hh) {
    float* oh = ob + hh * 128;
    atomicAdd(oh + lane,       hv0 * apca[hh]);
    atomicAdd(oh + lane + 32,  hv1 * apca[hh]);
    atomicAdd(oh + 64 + lane,  hv0 * api[hh]);
    atomicAdd(oh + 96 + lane,  hv1 * api[hh]);
  }
}

// ---------------------------------------------------------------------------
extern "C" void kernel_launch(void* const* d_in, const int* in_sizes, int n_in,
                              void* d_out, int out_size, void* d_ws, size_t ws_size,
                              hipStream_t stream) {
  const float* h_feat = (const float*)d_in[0];
  const float* pca    = (const float*)d_in[1];
  const float* pimg   = (const float*)d_in[2];
  const float* pca_w1 = (const float*)d_in[3];
  const float* pca_b1 = (const float*)d_in[4];
  const float* pca_w2 = (const float*)d_in[5];
  const float* pca_b2 = (const float*)d_in[6];
  const float* pi_w1  = (const float*)d_in[7];
  const float* pi_b1  = (const float*)d_in[8];
  const float* pi_w2  = (const float*)d_in[9];
  const float* pi_b2  = (const float*)d_in[10];
  const int*   src    = (const int*)d_in[11];
  const int*   dst    = (const int*)d_in[12];

  const int N = in_sizes[0] / 64;
  const int E = in_sizes[11];

  // ---- workspace carve (256B aligned) ----
  char* p = (char*)d_ws;
  auto carve = [&](size_t bytes) -> char* {
    char* r = p;
    p += (bytes + 255) & ~(size_t)255;
    return r;
  };
  float*    score_pca = (float*)   carve((size_t)E * 4 * sizeof(float));
  float*    score_pi  = (float*)   carve((size_t)E * 4 * sizeof(float));
  unsigned* nmax_pca  = (unsigned*)carve((size_t)N * 4 * sizeof(unsigned));
  unsigned* nmax_pi   = (unsigned*)carve((size_t)N * 4 * sizeof(unsigned));
  float*    nsum_pca  = (float*)   carve((size_t)N * 4 * sizeof(float));
  float*    nsum_pi   = (float*)   carve((size_t)N * 4 * sizeof(float));
  _Float16* w1f_pca   = (_Float16*)carve((size_t)128 * 128 * sizeof(_Float16));
  _Float16* w1f_pi    = (_Float16*)carve((size_t)64 * 128 * sizeof(_Float16));
  _Float16* w2f_pca   = (_Float16*)carve((size_t)128 * 16 * sizeof(_Float16));
  _Float16* w2f_pi    = (_Float16*)carve((size_t)128 * 16 * sizeof(_Float16));
  (void)ws_size; (void)n_in;

  // ---- zero accumulators + output (graph-capturable memset nodes) ----
  hipMemsetAsync(nmax_pca, 0, (size_t)N * 4 * sizeof(unsigned), stream);
  hipMemsetAsync(nmax_pi,  0, (size_t)N * 4 * sizeof(unsigned), stream);
  hipMemsetAsync(nsum_pca, 0, (size_t)N * 4 * sizeof(float), stream);
  hipMemsetAsync(nsum_pi,  0, (size_t)N * 4 * sizeof(float), stream);
  hipMemsetAsync(d_out,    0, (size_t)out_size * sizeof(float), stream);

  // ---- weight fragment prep (f32 -> f16 B-frag layout) ----
  prep_wfrag_kernel<<<4, 256, 0, stream>>>(pca_w1, 128, 128, w1f_pca, 8);  // 32 frags
  prep_wfrag_kernel<<<2, 256, 0, stream>>>(pi_w1,   64, 128, w1f_pi,  8);  // 16 frags
  prep_wfrag_kernel<<<1, 128, 0, stream>>>(pca_w2, 128,   4, w2f_pca, 1);  //  4 frags
  prep_wfrag_kernel<<<1, 128, 0, stream>>>(pi_w2,  128,   4, w2f_pi,  1);  //  4 frags

  // ---- stage 1: WMMA edge MLPs (1 wave = 16 edges, 8 waves / block) ----
  int tiles = (E + 15) >> 4;
  dim3 g1((tiles + 7) / 8);
  edge_mlp_kernel<4><<<g1, 256, 0, stream>>>(pca, 64, w1f_pca, pca_b1, w2f_pca,
                                             pca_b2, src, dst, score_pca, nmax_pca, E);
  edge_mlp_kernel<2><<<g1, 256, 0, stream>>>(pimg, 32, w1f_pi, pi_b1, w2f_pi,
                                             pi_b2, src, dst, score_pi, nmax_pi, E);

  // ---- stage 2: edge softmax (exp in place + node sums) ----
  int T = E * 4;
  softmax_expsum_kernel<<<(T + 255) / 256, 256, 0, stream>>>(score_pca, dst, nmax_pca,
                                                             nsum_pca, E);
  softmax_expsum_kernel<<<(T + 255) / 256, 256, 0, stream>>>(score_pi, dst, nmax_pi,
                                                             nsum_pi, E);

  // ---- stage 3: fused aggregation (1 wave per edge) ----
  aggregate_kernel<<<(E + 7) / 8, 256, 0, stream>>>(h_feat, score_pca, nsum_pca,
                                                    score_pi, nsum_pi, src, dst,
                                                    (float*)d_out, E);
}
